// HolographicBlock_55783035240635
// MI455X (gfx1250) — compile-verified, matrix operand order checked by reading, not run on hardware
//
#include <hip/hip_runtime.h>

// ---------------- problem constants (from reference) ----------------
#define Bb   2
#define Ss   2048
#define Dd   1024
#define Rr   64
#define Hh   8
#define HDd  128
#define HALF 64
#define Ntok (Bb * Ss)
#define EPSV 1.1920929e-07f

// gfx1250 async global->LDS copy, if this toolchain declares it
#if defined(__has_builtin)
#if __has_builtin(__builtin_amdgcn_global_load_async_to_lds_b128) && \
    __has_builtin(__builtin_amdgcn_s_wait_asynccnt)
#define HAVE_ASYNC_LDS 1
#endif
#endif
#ifndef HAVE_ASYNC_LDS
#define HAVE_ASYNC_LDS 0
#endif

typedef __attribute__((ext_vector_type(16))) __bf16 v16bf;
typedef __attribute__((ext_vector_type(8)))  __bf16 v8bf;
typedef __attribute__((ext_vector_type(8)))  float  v8f;
typedef __attribute__((ext_vector_type(4)))  int    v4i;

#if HAVE_ASYNC_LDS
typedef __attribute__((address_space(1))) v4i* gl_v4i_ptr;
typedef __attribute__((address_space(3))) v4i* lds_v4i_ptr;
#endif

// ---------------- WMMA fragment load helpers ----------------
// 16-bit A-operand layout (ISA 7.12.2): lane L (L<16) holds row L,
// k = 0..7 and 16..23; lane L+16 holds row L, k = 8..15 and 24..31.
// Same pattern serves the B-operand when B is stored column-major.
__device__ __forceinline__ v16bf load_frag(const __bf16* base, int ld, int k0, int lane) {
  int r   = lane & 15;
  int sel = (lane >> 4) & 1;
  const __bf16* p = base + (size_t)r * ld + k0 + sel * 8;
  v8bf lo = *reinterpret_cast<const v8bf*>(p);
  v8bf hi = *reinterpret_cast<const v8bf*>(p + 16);
  v16bf o;
#pragma unroll
  for (int i = 0; i < 8; i++) { o[i] = lo[i]; o[i + 8] = hi[i]; }
  return o;
}

__device__ __forceinline__ v8f wmma_bf16(v16bf a, v16bf b, v8f c) {
  return __builtin_amdgcn_wmma_f32_16x16x32_bf16(false, a, false, b, (short)0, c,
                                                 false, false);
}

// ---------------- prep: transposed / pre-scaled bf16 basis ----------------
__global__ __launch_bounds__(256) void prep_kernel(
    const float* __restrict__ A, const float* __restrict__ Bm,
    const float* __restrict__ Cq, const float* __restrict__ Ck,
    const float* __restrict__ Cv, const float* __restrict__ Co,
    const float* __restrict__ Cfc, const float* __restrict__ Cp,
    __bf16* __restrict__ Acm, __bf16* __restrict__ B6) {
  int j = blockIdx.x * blockDim.x + threadIdx.x;
  if (j >= Rr * Dd) return;
  // Acm[r*D + d] = basis_A[d*R + r]   (column-major for WMMA B-operand)
  int r = j / Dd, d = j % Dd;
  Acm[(size_t)r * Dd + d] = (__bf16)A[(size_t)d * Rr + r];
  // B6[i][d*R + r2] = C_i[r2] * basis_B[r2*D + d]
  int d2 = j / Rr, r2 = j % Rr;
  float bv = Bm[(size_t)r2 * Dd + d2];
  const size_t sz = (size_t)Rr * Dd;
  B6[0 * sz + j] = (__bf16)(bv * Cq[r2]);
  B6[1 * sz + j] = (__bf16)(bv * Ck[r2]);
  B6[2 * sz + j] = (__bf16)(bv * Cv[r2]);
  B6[3 * sz + j] = (__bf16)(bv * Co[r2]);
  B6[4 * sz + j] = (__bf16)(bv * Cfc[r2]);
  B6[5 * sz + j] = (__bf16)(bv * Cp[r2]);
}

// ---------------- RMSNorm -> bf16 ----------------
__global__ __launch_bounds__(256) void rms_kernel(const float* __restrict__ x,
                                                  const float* __restrict__ scale,
                                                  __bf16* __restrict__ out) {
  __shared__ float red[256];
  int t = blockIdx.x;
  const float* row = x + (size_t)t * Dd;
  float ss = 0.f;
  for (int i = threadIdx.x; i < Dd; i += 256) { float v = row[i]; ss += v * v; }
  red[threadIdx.x] = ss;
  __syncthreads();
  for (int s = 128; s > 0; s >>= 1) {
    if (threadIdx.x < s) red[threadIdx.x] += red[threadIdx.x + s];
    __syncthreads();
  }
  float inv = rsqrtf(red[0] * (1.0f / Dd) + EPSV);
  for (int i = threadIdx.x; i < Dd; i += 256)
    out[(size_t)t * Dd + i] = (__bf16)(row[i] * inv * scale[i]);
}

// ---------------- generic bf16 WMMA GEMM ----------------
// C(MxN) = A(MxK, row-major bf16) @ Bcm(N columns, each K contiguous bf16)
// One wave -> 16x64 tile.  The 8 waves of a block share the same 64 output
// columns, so for K==64 the whole 8KB B panel is staged once in LDS
// (async global->LDS when the toolchain has the gfx1250 builtin).  For
// larger K the loop is unrolled by 2 with ping-pong register buffers so
// loads overlap matrix math with no copy movs.  Epilogue modes:
//  0: bf16 row-major          1: bf16 scatter to (b,h,s,hd)
//  2: bf16 scatter to (b,h,hd,s)   3: fp32 out = resid + v
//  4: bf16 silu(v) row-major  5: fp32 out += v
__global__ __launch_bounds__(256) void gemm_kernel(
    const __bf16* __restrict__ A, const __bf16* __restrict__ Bcm, int M, int N,
    int K, __bf16* __restrict__ outb, float* __restrict__ outf,
    const float* __restrict__ resid, int mode) {
  __shared__ __align__(16) __bf16 bpanel[64 * 64];  // 8 KB, used when K==64
  int lane = threadIdx.x & 31;
  int wave = threadIdx.x >> 5;
  int wg = blockIdx.x * 8 + wave;
  int mtiles = M >> 4;
  int mt = wg % mtiles, nt = wg / mtiles;
  int row0 = mt * 16, col0 = nt * 64;

  v8f acc[4];
#pragma unroll
  for (int c = 0; c < 4; c++)
#pragma unroll
    for (int j = 0; j < 8; j++) acc[c][j] = 0.f;

  const __bf16* Abase = A + (size_t)row0 * K;

  if (K == 64) {
    // ---- stage the contiguous 8KB B panel (64 cols x 64 k) in LDS ----
    const __bf16* src = Bcm + (size_t)col0 * 64;
#if HAVE_ASYNC_LDS
    for (int i = threadIdx.x; i < 512; i += 256)
      __builtin_amdgcn_global_load_async_to_lds_b128(
          (gl_v4i_ptr)(size_t)(src + (size_t)i * 8),
          (lds_v4i_ptr)(bpanel + (size_t)i * 8), 0, 0);
    __builtin_amdgcn_s_wait_asynccnt(0);
    __syncthreads();
#else
    for (int i = threadIdx.x; i < 512; i += 256)
      *reinterpret_cast<v8bf*>(bpanel + (size_t)i * 8) =
          *reinterpret_cast<const v8bf*>(src + (size_t)i * 8);
    __syncthreads();
#endif
    v16bf af0 = load_frag(Abase, 64, 0, lane);
    v16bf af1 = load_frag(Abase, 64, 32, lane);
#pragma unroll
    for (int c = 0; c < 4; c++) {
      v16bf b0 = load_frag(bpanel + (size_t)(c * 16) * 64, 64, 0, lane);
      acc[c] = wmma_bf16(af0, b0, acc[c]);
    }
#pragma unroll
    for (int c = 0; c < 4; c++) {
      v16bf b1 = load_frag(bpanel + (size_t)(c * 16) * 64, 64, 32, lane);
      acc[c] = wmma_bf16(af1, b1, acc[c]);
    }
  } else {
    // ---- ping-pong pipelined global path (iters = K/32 is even) ----
    const __bf16* Bcol[4];
#pragma unroll
    for (int c = 0; c < 4; c++) Bcol[c] = Bcm + (size_t)(col0 + c * 16) * K;
    int iters = K >> 5;
    v16bf afA = load_frag(Abase, K, 0, lane);
    v16bf bfA[4];
#pragma unroll
    for (int c = 0; c < 4; c++) bfA[c] = load_frag(Bcol[c], K, 0, lane);

    for (int it = 0; it < iters; it += 2) {
      int k1 = (it + 1 < iters) ? (it + 1) * 32 : 0;
      v16bf afB = load_frag(Abase, K, k1, lane);
      v16bf bfB[4];
#pragma unroll
      for (int c = 0; c < 4; c++) bfB[c] = load_frag(Bcol[c], K, k1, lane);
      int kp = (it + 8 < iters ? it + 8 : it) * 32;
      __builtin_prefetch(Abase + (size_t)(lane & 15) * K + kp, 0, 3);
#pragma unroll
      for (int c = 0; c < 4; c++) acc[c] = wmma_bf16(afA, bfA[c], acc[c]);

      int k2 = (it + 2 < iters) ? (it + 2) * 32 : 0;
      afA = load_frag(Abase, K, k2, lane);
#pragma unroll
      for (int c = 0; c < 4; c++) bfA[c] = load_frag(Bcol[c], K, k2, lane);
#pragma unroll
      for (int c = 0; c < 4; c++) acc[c] = wmma_bf16(afB, bfB[c], acc[c]);
    }
  }

  int n = lane & 15, hsel = lane >> 4;
#pragma unroll
  for (int c = 0; c < 4; c++) {
#pragma unroll
    for (int j = 0; j < 8; j++) {
      int m = hsel * 8 + j;
      int r = row0 + m;
      int col = col0 + c * 16 + n;
      float v = acc[c][j];
      if (mode == 0) {
        outb[(size_t)r * N + col] = (__bf16)v;
      } else if (mode == 1 || mode == 2) {
        int b = r / Ss, s2 = r % Ss;
        int h = col >> 7, hd = col & (HDd - 1);
        size_t idx = (mode == 1)
                         ? ((size_t)((b * Hh + h) * Ss + s2) * HDd + hd)
                         : ((size_t)((b * Hh + h) * HDd + hd) * Ss + s2);
        outb[idx] = (__bf16)v;
      } else if (mode == 3) {
        size_t idx = (size_t)r * N + col;
        outf[idx] = resid[idx] + v;
      } else if (mode == 4) {
        float sv = v / (1.f + __expf(-v));
        outb[(size_t)r * N + col] = (__bf16)sv;
      } else {  // 5
        size_t idx = (size_t)r * N + col;
        outf[idx] += v;
      }
    }
  }
}

// ---------------- RoPE in place on q,k (layout b,h,s,hd) ----------------
__global__ __launch_bounds__(256) void rope_kernel(__bf16* __restrict__ q,
                                                   __bf16* __restrict__ k) {
  int t = blockIdx.x * blockDim.x + threadIdx.x;  // B*H*S*HALF threads, exact
  int i = t % HALF;
  int s = (t / HALF) % Ss;
  int bh = t / (HALF * Ss);
  float invf = __expf(-((float)(2 * i) / (float)HDd) * 9.210340371976184f);
  float ang = (float)s * invf;
  float c = __cosf(ang), sn = __sinf(ang);
  size_t base = ((size_t)bh * Ss + s) * HDd;
  float q1 = (float)q[base + i], q2 = (float)q[base + i + HALF];
  q[base + i]        = (__bf16)(q1 * c + q2 * sn);
  q[base + i + HALF] = (__bf16)(-q1 * sn + q2 * c);
  float k1 = (float)k[base + i], k2 = (float)k[base + i + HALF];
  k[base + i]        = (__bf16)(k1 * c + k2 * sn);
  k[base + i + HALF] = (__bf16)(-k1 * sn + k2 * c);
}

// ---------------- flash-style causal attention (pipelined) ----------------
// One wave per 16-query tile. q,k: (b,h,s,hd) bf16; vt: (b,h,hd,s) bf16;
// y out: (b,s,h*hd) bf16.  Next 32-key tile's K fragments are preloaded so
// the softmax VALU work hides their latency.
__global__ __launch_bounds__(256) void attn_kernel(const __bf16* __restrict__ q,
                                                   const __bf16* __restrict__ k,
                                                   const __bf16* __restrict__ vt,
                                                   __bf16* __restrict__ y) {
  __shared__ __align__(16) __bf16 plds[8][16][32];
  int lane = threadIdx.x & 31;
  int wave = threadIdx.x >> 5;
  int wg = blockIdx.x * 8 + wave;          // 0..2047
  const int qtiles = Ss / 16;              // 128
  int qt = wg % qtiles;
  int bh = wg / qtiles;                    // b*H + h
  int q0 = qt * 16;
  const __bf16* qrow = q + (size_t)bh * Ss * HDd;
  const __bf16* krow = k + (size_t)bh * Ss * HDd;
  const __bf16* vrow = vt + (size_t)bh * HDd * Ss;

  v16bf qf[4];
#pragma unroll
  for (int kk = 0; kk < 4; kk++)
    qf[kk] = load_frag(qrow + (size_t)q0 * HDd, HDd, kk * 32, lane);

  v8f acc[8];
  float mrun[8], lrun[8];
#pragma unroll
  for (int s = 0; s < 8; s++)
#pragma unroll
    for (int j = 0; j < 8; j++) acc[s][j] = 0.f;
#pragma unroll
  for (int j = 0; j < 8; j++) { mrun[j] = -3.0e38f; lrun[j] = 0.f; }

  const float sc = 0.08838834764831845f;  // 1/sqrt(128)
  int n = lane & 15, hsel = lane >> 4;
  int nkv = q0 / 32 + 1;

  // prologue: K fragments for the first key tile
  v16bf kf0[4], kf1[4];
#pragma unroll
  for (int kk = 0; kk < 4; kk++) {
    kf0[kk] = load_frag(krow, HDd, kk * 32, lane);
    kf1[kk] = load_frag(krow + (size_t)16 * HDd, HDd, kk * 32, lane);
  }

  for (int it = 0; it < nkv; ++it) {
    int j0 = it * 32;
    v8f s0, s1;
#pragma unroll
    for (int j = 0; j < 8; j++) { s0[j] = 0.f; s1[j] = 0.f; }
#pragma unroll
    for (int kk = 0; kk < 4; kk++) {
      s0 = wmma_bf16(qf[kk], kf0[kk], s0);
      s1 = wmma_bf16(qf[kk], kf1[kk], s1);
    }
    // preload next key tile's fragments; softmax below hides the latency
    int jn = (it + 1 < nkv) ? (it + 1) * 32 : 0;
#pragma unroll
    for (int kk = 0; kk < 4; kk++) {
      kf0[kk] = load_frag(krow + (size_t)jn * HDd, HDd, kk * 32, lane);
      kf1[kk] = load_frag(krow + (size_t)(jn + 16) * HDd, HDd, kk * 32, lane);
    }
    // prefetch the V panel rows for this key tile toward the WGP
    __builtin_prefetch(vrow + (size_t)(lane & 15) * 16 * Ss + j0, 0, 3);

    float alpha[8], pa[8], pb[8];
#pragma unroll
    for (int j = 0; j < 8; j++) {
      int m = hsel * 8 + j;
      float a = s0[j] * sc;
      float b = s1[j] * sc;
      if (j0 + n > q0 + m) a = -3.0e38f;
      if (j0 + 16 + n > q0 + m) b = -3.0e38f;
      float mx = fmaxf(a, b);
#pragma unroll
      for (int off = 1; off < 16; off <<= 1)
        mx = fmaxf(mx, __shfl_xor(mx, off, 32));
      float mnew = fmaxf(mrun[j], mx);
      alpha[j] = __expf(mrun[j] - mnew);
      a = __expf(a - mnew);
      b = __expf(b - mnew);
      float rs = a + b;
#pragma unroll
      for (int off = 1; off < 16; off <<= 1) rs += __shfl_xor(rs, off, 32);
      lrun[j] = lrun[j] * alpha[j] + rs;
      mrun[j] = mnew;
      pa[j] = a;
      pb[j] = b;
    }
#pragma unroll
    for (int s = 0; s < 8; s++)
#pragma unroll
      for (int j = 0; j < 8; j++) acc[s][j] *= alpha[j];

    // accumulator layout -> A-fragment layout through wave-private LDS
#pragma unroll
    for (int j = 0; j < 8; j++) {
      int m = hsel * 8 + j;
      plds[wave][m][n] = (__bf16)pa[j];
      plds[wave][m][16 + n] = (__bf16)pb[j];
    }
    v16bf pf;
    {
      const __bf16* pp = &plds[wave][lane & 15][hsel * 8];
#pragma unroll
      for (int i = 0; i < 8; i++) { pf[i] = pp[i]; pf[i + 8] = pp[16 + i]; }
    }
#pragma unroll
    for (int s = 0; s < 8; s++) {
      v16bf vf = load_frag(vrow + (size_t)(s * 16) * Ss, Ss, j0, lane);
      acc[s] = wmma_bf16(pf, vf, acc[s]);
    }
  }

  int b = bh / Hh, h = bh % Hh;
#pragma unroll
  for (int s = 0; s < 8; s++) {
#pragma unroll
    for (int j = 0; j < 8; j++) {
      int m = hsel * 8 + j;
      float yv = acc[s][j] / lrun[j];
      size_t tok = (size_t)b * Ss + q0 + m;
      y[tok * Dd + h * HDd + s * 16 + n] = (__bf16)yv;
    }
  }
}

// ---------------- host orchestration ----------------
extern "C" void kernel_launch(void* const* d_in, const int* in_sizes, int n_in,
                              void* d_out, int out_size, void* d_ws,
                              size_t ws_size, hipStream_t stream) {
  const float* x      = (const float*)d_in[0];
  const float* bA     = (const float*)d_in[1];
  const float* bB     = (const float*)d_in[2];
  const float* Cq     = (const float*)d_in[3];
  const float* Ck     = (const float*)d_in[4];
  const float* Cv     = (const float*)d_in[5];
  const float* Co     = (const float*)d_in[6];
  const float* Cfc    = (const float*)d_in[7];
  const float* Cp     = (const float*)d_in[8];
  const float* scale1 = (const float*)d_in[9];
  const float* scale2 = (const float*)d_in[10];
  float* out = (float*)d_out;

  char* ws = (char*)d_ws;
  const size_t szND = (size_t)Ntok * Dd * sizeof(__bf16);   // 8 MB
  const size_t szNR = (size_t)Ntok * Rr * sizeof(__bf16);   // 512 KB
  const size_t szRD = (size_t)Rr * Dd * sizeof(__bf16);     // 128 KB
  __bf16* nbuf = (__bf16*)(ws);
  __bf16* Tbuf = (__bf16*)(ws + szND);
  __bf16* qbuf = (__bf16*)(ws + szND + szNR);               // also reused for m
  __bf16* kbuf = (__bf16*)(ws + 2 * szND + szNR);
  __bf16* vtb  = (__bf16*)(ws + 3 * szND + szNR);
  __bf16* ybuf = (__bf16*)(ws + 4 * szND + szNR);
  __bf16* Acm  = (__bf16*)(ws + 5 * szND + szNR);
  __bf16* B6   = (__bf16*)(ws + 5 * szND + szNR + szRD);
  __bf16* Bq = B6 + 0 * (size_t)Rr * Dd;
  __bf16* Bk = B6 + 1 * (size_t)Rr * Dd;
  __bf16* Bv = B6 + 2 * (size_t)Rr * Dd;
  __bf16* Bo = B6 + 3 * (size_t)Rr * Dd;
  __bf16* Bf = B6 + 4 * (size_t)Rr * Dd;
  __bf16* Bp = B6 + 5 * (size_t)Rr * Dd;

  auto gemm = [&](const __bf16* A, const __bf16* Bm, int M, int N, int K,
                  __bf16* outb, float* outf, const float* resid, int mode) {
    int blocks = (M / 16) * (N / 64) / 8;
    gemm_kernel<<<blocks, 256, 0, stream>>>(A, Bm, M, N, K, outb, outf, resid,
                                            mode);
  };

  prep_kernel<<<(Rr * Dd + 255) / 256, 256, 0, stream>>>(bA, bB, Cq, Ck, Cv, Co,
                                                         Cfc, Cp, Acm, B6);
  // n1 = rms(x, scale1)
  rms_kernel<<<Ntok, 256, 0, stream>>>(x, scale1, nbuf);
  // T = n1 @ A ; q/k/v = T @ (C*B)  (C folded into Bq/Bk/Bv)
  gemm(nbuf, Acm, Ntok, Rr, Dd, Tbuf, nullptr, nullptr, 0);
  gemm(Tbuf, Bq, Ntok, Dd, Rr, qbuf, nullptr, nullptr, 1);
  gemm(Tbuf, Bk, Ntok, Dd, Rr, kbuf, nullptr, nullptr, 1);
  gemm(Tbuf, Bv, Ntok, Dd, Rr, vtb, nullptr, nullptr, 2);
  rope_kernel<<<(Bb * Hh * Ss * HALF) / 256, 256, 0, stream>>>(qbuf, kbuf);
  attn_kernel<<<(Bb * Hh * (Ss / 16)) / 8, 256, 0, stream>>>(qbuf, kbuf, vtb,
                                                             ybuf);
  // x2 = x + blin(y, C_o)
  gemm(ybuf, Acm, Ntok, Rr, Dd, Tbuf, nullptr, nullptr, 0);
  gemm(Tbuf, Bo, Ntok, Dd, Rr, nullptr, out, x, 3);
  // n2 = rms(x2, scale2); m = silu(blin(n2, C_fc)); out = x2 + blin(m, C_proj)
  rms_kernel<<<Ntok, 256, 0, stream>>>(out, scale2, nbuf);
  gemm(nbuf, Acm, Ntok, Rr, Dd, Tbuf, nullptr, nullptr, 0);
  gemm(Tbuf, Bf, Ntok, Dd, Rr, qbuf, nullptr, nullptr, 4);
  gemm(qbuf, Acm, Ntok, Rr, Dd, Tbuf, nullptr, nullptr, 0);
  gemm(Tbuf, Bp, Ntok, Dd, Rr, nullptr, out, nullptr, 5);
  (void)in_sizes; (void)n_in; (void)out_size; (void)ws_size;
}
